// Reverb_90142773608909
// MI455X (gfx1250) — matrix operand order repaired and checked by reading, not compiled, not run on hardware
//
#include <hip/hip_runtime.h>

// ---------------- problem constants (from reference) ----------------
constexpr int N_AUDIO = 44100 * 60;            // 2,646,000 samples
constexpr int M_IR    = 220500;                // IR taps
constexpr int MB2     = (M_IR + 15) / 16;      // 13,782 tap blocks (16 taps each)
constexpr int HP      = MB2 * 16;              // 220,512 padded IR length
constexpr int TILES   = (N_AUDIO + 255) / 256; // 10,336 tiles of 256 outputs
constexpr int PADN    = 2866688;               // padded x length (covers max index)
constexpr int CHUNK   = 16;                    // tap blocks per LDS stage
constexpr int ROUNDS  = (MB2 + CHUNK - 1) / CHUNK; // 862
constexpr int MB3     = ROUNDS * CHUNK;        // 13,792 (table padded w/ zeros)
constexpr int CBYTES  = 2048;                  // A-table bytes per tap block (Th+Tl)
constexpr int STAGE_BYTES = CHUNK * CBYTES;    // 32,768

typedef _Float16 v8h  __attribute__((ext_vector_type(8)));
typedef _Float16 v16h __attribute__((ext_vector_type(16)));
typedef float    v8f  __attribute__((ext_vector_type(8)));

// ---------------- workspace layout (bytes) ----------------
constexpr size_t WS_XH   = 256;                          // PADN f16 (x hi part)
constexpr size_t WS_XL   = WS_XH + (size_t)PADN * 2;     // PADN f16 (x lo part)
constexpr size_t WS_HPAD = WS_XL + (size_t)PADN * 2;     // HP f32 (padded IR)
constexpr size_t WS_ATAB = WS_HPAD + (size_t)HP * 4;     // MB3*2048 A-operand table
// total ~40.6 MB

// ---------------- kernel A: zero-pad + hi/lo f16 split of x; pad h ----------------
__global__ __launch_bounds__(256) void reverb_pad_kernel(
    const float* __restrict__ audio, const float* __restrict__ ir,
    _Float16* __restrict__ xh, _Float16* __restrict__ xl,
    float* __restrict__ hpad, unsigned* __restrict__ maxslot)
{
    int idx = blockIdx.x * 256 + threadIdx.x;
    if (idx < PADN) {
        float v = (idx < N_AUDIO) ? audio[idx] : 0.0f;
        _Float16 h = (_Float16)v;
        xh[idx] = h;
        xl[idx] = (_Float16)(v - (float)h);
    }
    if (idx < HP) hpad[idx] = (idx < M_IR) ? ir[idx] : 0.0f;
    if (idx == 0) *maxslot = 0u;
}

// ---------------- kernel B: build the Toeplitz A-operand table ----------------
// For tap block c, the 16x32 strip T[i][K] = h[16c + K - i] if 0 <= K-i < 16 else 0
// (merged lower/upper block-Toeplitz; K=32 spans the 2-block spill of the FIR).
// Stored in a contiguous-k A-operand layout (same k order as the B operand):
//   lane l: row i = l&15, hi = l>>4; vector element e holds K = e + 16*hi.
// Memory layout per c (2048 B): part-major, 16 B per lane per part -> conflict-free
//   ds_load_b128:  Th part0 @0, Th part1 @512, Tl part0 @1024, Tl part1 @1536.
__global__ __launch_bounds__(256) void reverb_atab_kernel(
    const float* __restrict__ hpad, char* __restrict__ atab)
{
    int tid  = blockIdx.x * 256 + threadIdx.x;
    int c    = tid >> 5;
    int lane = tid & 31;
    if (c >= MB3) return;
    int i  = lane & 15;
    int hi = lane >> 4;

    v8h th[2], tl[2];
#pragma unroll
    for (int t = 0; t < 2; ++t) {
#pragma unroll
        for (int p = 0; p < 8; ++p) {
            int e = t * 8 + p;
            int K = e + 16 * hi;
            int d = K - i;
            float v = (c < MB2 && d >= 0 && d < 16) ? hpad[16 * c + d] : 0.0f;
            _Float16 h = (_Float16)v;
            th[t][p] = h;
            tl[t][p] = (_Float16)(v - (float)h);
        }
    }
    char* dst = atab + (size_t)c * CBYTES;
    *reinterpret_cast<v8h*>(dst +    0 + lane * 16) = th[0];
    *reinterpret_cast<v8h*>(dst +  512 + lane * 16) = th[1];
    *reinterpret_cast<v8h*>(dst + 1024 + lane * 16) = tl[0];
    *reinterpret_cast<v8h*>(dst + 1536 + lane * 16) = tl[1];
}

// ---------------- kernel C: split-f16 block-Toeplitz FIR via WMMA ----------------
// One wave computes outputs [256*tile, 256*tile+256): Y[i][j] = y[16*(b0+j)+i].
// D ~= sum_c ( Th_c*Xh + Th_c*Xl + Tl_c*Xh ), f32 accumulate (split-f16 precision).
__global__ __launch_bounds__(256) void reverb_wmma_kernel(
    const _Float16* __restrict__ xh, const _Float16* __restrict__ xl,
    const char* __restrict__ atab, float* __restrict__ out,
    unsigned* __restrict__ maxslot)
{
    __shared__ __align__(16) char smem[2 * STAGE_BYTES];   // 64 KB double buffer

    const int tid  = threadIdx.x;
    const int lane = tid & 31;
    const int wave = tid >> 5;
    const int tile = blockIdx.x * 8 + wave;   // grid sized so tile < TILES
    const int col  = lane & 15;               // B/D column j
    const int hi   = lane >> 4;
    const int b0   = tile * 16;

    // B operand: lane j holds K = 16*hi + e of column j:
    //   Xext[K][j] = x[16*(b0+c+j) + K]  -> 16 consecutive halves per lane.
    const _Float16* xhb = xh + 16 * (b0 + col) + 16 * hi;
    const _Float16* xlb = xl + 16 * (b0 + col) + 16 * hi;

    v8f aA0 = {}, aA1 = {}, aA2 = {};   // even-cc accumulators
    v8f aB0 = {}, aB1 = {}, aB2 = {};   // odd-cc accumulators

    // prologue: stage round 0 (128 B per thread, coalesced b128)
    {
        const int4* g = reinterpret_cast<const int4*>(atab);
        int4* s = reinterpret_cast<int4*>(smem);
#pragma unroll
        for (int k = 0; k < 8; ++k) s[tid + k * 256] = g[tid + k * 256];
    }
    __syncthreads();

    for (int r = 0; r < ROUNDS; ++r) {
        const int cur = r & 1;
        if (r + 1 < ROUNDS) {   // prefetch next stage into the other buffer
            const int4* g = reinterpret_cast<const int4*>(atab + (size_t)(r + 1) * STAGE_BYTES);
            int4* s = reinterpret_cast<int4*>(smem + (cur ^ 1) * STAGE_BYTES);
#pragma unroll
            for (int k = 0; k < 8; ++k) s[tid + k * 256] = g[tid + k * 256];
        }

        const char* sbase = smem + cur * STAGE_BYTES + lane * 16;
        const _Float16* xhc = xhb + (size_t)r * CHUNK * 16;
        const _Float16* xlc = xlb + (size_t)r * CHUNK * 16;

#pragma unroll
        for (int cc = 0; cc < CHUNK; ++cc) {
            const char* sA = sbase + cc * CBYTES;
            v8h th0 = *reinterpret_cast<const v8h*>(sA);
            v8h th1 = *reinterpret_cast<const v8h*>(sA + 512);
            v8h tl0 = *reinterpret_cast<const v8h*>(sA + 1024);
            v8h tl1 = *reinterpret_cast<const v8h*>(sA + 1536);
            v16h th = __builtin_shufflevector(th0, th1, 0,1,2,3,4,5,6,7,8,9,10,11,12,13,14,15);
            v16h tl = __builtin_shufflevector(tl0, tl1, 0,1,2,3,4,5,6,7,8,9,10,11,12,13,14,15);

            v16h bh = *reinterpret_cast<const v16h*>(xhc + cc * 16);
            v16h bl = *reinterpret_cast<const v16h*>(xlc + cc * 16);

            if (cc & 1) {
                aB0 = __builtin_amdgcn_wmma_f32_16x16x32_f16(false, th, false, bh, (short)0, aB0, false, false);
                aB1 = __builtin_amdgcn_wmma_f32_16x16x32_f16(false, th, false, bl, (short)0, aB1, false, false);
                aB2 = __builtin_amdgcn_wmma_f32_16x16x32_f16(false, tl, false, bh, (short)0, aB2, false, false);
            } else {
                aA0 = __builtin_amdgcn_wmma_f32_16x16x32_f16(false, th, false, bh, (short)0, aA0, false, false);
                aA1 = __builtin_amdgcn_wmma_f32_16x16x32_f16(false, th, false, bl, (short)0, aA1, false, false);
                aA2 = __builtin_amdgcn_wmma_f32_16x16x32_f16(false, tl, false, bh, (short)0, aA2, false, false);
            }
        }
        __syncthreads();
    }

    v8f acc = (aA0 + aB0) + (aA1 + aB1) + (aA2 + aB2);

    // D layout: VGPR r -> row r + 8*hi, column = lane&15 -> 8 consecutive outputs per lane.
    const int n_base = 256 * tile + 16 * col + 8 * hi;
    float m = 0.0f;
#pragma unroll
    for (int rr = 0; rr < 8; ++rr) {
        int n = n_base + rr;
        if (n < N_AUDIO) {
            out[n] = acc[rr];
            m = fmaxf(m, fabsf(acc[rr]));
        }
    }
    // wave32 max reduction + one atomic (uint order == float order for values >= 0)
#pragma unroll
    for (int off = 16; off >= 1; off >>= 1)
        m = fmaxf(m, __shfl_xor(m, off, 32));
    if (lane == 0)
        atomicMax(maxslot, __float_as_uint(m));
}

// ---------------- kernel D: global max-abs normalization ----------------
__global__ __launch_bounds__(256) void reverb_norm_kernel(
    float* __restrict__ out, const unsigned* __restrict__ maxslot)
{
    int idx = blockIdx.x * 256 + threadIdx.x;
    float mv = __uint_as_float(*maxslot);
    if (idx < N_AUDIO) out[idx] = out[idx] / mv;
}

// ---------------- host launch ----------------
extern "C" void kernel_launch(void* const* d_in, const int* in_sizes, int n_in,
                              void* d_out, int out_size, void* d_ws, size_t ws_size,
                              hipStream_t stream) {
    (void)in_sizes; (void)n_in; (void)out_size; (void)ws_size;
    const float* audio = reinterpret_cast<const float*>(d_in[0]);
    const float* ir    = reinterpret_cast<const float*>(d_in[1]);
    float*       out   = reinterpret_cast<float*>(d_out);

    char*      ws      = reinterpret_cast<char*>(d_ws);
    unsigned*  maxslot = reinterpret_cast<unsigned*>(ws);
    _Float16*  xh      = reinterpret_cast<_Float16*>(ws + WS_XH);
    _Float16*  xl      = reinterpret_cast<_Float16*>(ws + WS_XL);
    float*     hpad    = reinterpret_cast<float*>(ws + WS_HPAD);
    char*      atab    = ws + WS_ATAB;

    reverb_pad_kernel <<<PADN / 256, 256, 0, stream>>>(audio, ir, xh, xl, hpad, maxslot);
    reverb_atab_kernel<<<(MB3 * 32) / 256, 256, 0, stream>>>(hpad, atab);
    reverb_wmma_kernel<<<TILES / 8, 256, 0, stream>>>(xh, xl, atab, out, maxslot);
    reverb_norm_kernel<<<(N_AUDIO + 255) / 256, 256, 0, stream>>>(out, maxslot);
}
